// VisionMamba_29583734735074
// MI455X (gfx1250) — matrix operand order repaired and checked by reading, not compiled, not run on hardware
//
#include <hip/hip_runtime.h>
#include <hip/hip_bf16.h>

// ---------------------------------------------------------------------------
// Vision Mamba forward for MI455X (gfx1250), wave32 + WMMA.
// All GEMMs run through one v_wmma_f32_16x16x32_f16 kernel; the selective
// scan is a register-resident recurrence, parallel over (batch, d_inner).
// ---------------------------------------------------------------------------

typedef _Float16 v16h_t __attribute__((ext_vector_type(16)));
typedef float    v8f_t  __attribute__((ext_vector_type(8)));

#define MROWS 6272          // B * L = 8 * 784
#define LSEQ  784
#define NB    8
#define DIN   392
#define DMODEL 196

__device__ __forceinline__ float silu_f(float x) {
    return x / (1.0f + __expf(-x));
}

// ---------------------------------------------------------------------------
// Generic WMMA GEMM:  C(M x N) = A(M x Kpad) * W(N x Kpad)^T  [+ bias][act]
// A, W are f16 row-major with K padded to a multiple of 32 (pad = 0).
// M is always a multiple of 16 (6272).  One wave computes one 16x16 tile.
// blockDim = (32, 4): 4 waves per block, each owns a distinct N tile.
// act: 0 = none, 1 = silu, 2 = softplus
// ---------------------------------------------------------------------------
__global__ void wmma_gemm_kernel(const _Float16* __restrict__ A,
                                 const _Float16* __restrict__ W,
                                 const float* __restrict__ bias,
                                 float* __restrict__ C,
                                 int Kpad, int N, int Ntiles, int ldc, int act) {
    const int lane = threadIdx.x;          // 0..31
    const int nt   = blockIdx.y * 4 + threadIdx.y;
    if (nt >= Ntiles) return;
    const int mt   = blockIdx.x;
    const int l15  = lane & 15;
    const int half = lane >> 4;

    const _Float16* __restrict__ arow = A + (size_t)(mt * 16 + l15) * Kpad;
    const _Float16* __restrict__ brow = W + (size_t)(nt * 16 + l15) * Kpad;

    v8f_t acc = {0.f, 0.f, 0.f, 0.f, 0.f, 0.f, 0.f, 0.f};

    for (int k0 = 0; k0 < Kpad; k0 += 32) {
        v16h_t af, bf;
        // A fragment: 16x32 f16 (ISA 7.12.2): VGPR v holds K-pair
        //   k = ((v&4)?16:0) + half*8 + (v&3)*2
        // B fragment: 32x16, per-lane column n=l15, K packed 2/VGPR:
        //   k = ((v&4)?16:0) + half*8 + (v&3)*2
#pragma unroll
        for (int v = 0; v < 8; ++v) {
            int ka = k0 + ((v & 4) ? 16 : 0) + half * 8 + (v & 3) * 2;
            af[2 * v]     = arow[ka];
            af[2 * v + 1] = arow[ka + 1];
            int kb = k0 + ((v & 4) ? 16 : 0) + half * 8 + (v & 3) * 2;
            bf[2 * v]     = brow[kb];
            bf[2 * v + 1] = brow[kb + 1];
        }
        acc = __builtin_amdgcn_wmma_f32_16x16x32_f16(
            /*neg_a=*/false, af, /*neg_b=*/false, bf,
            /*c_mod=*/(short)0, acc, /*reuse_a=*/false, /*reuse_b=*/false);
    }

    const int ncol = nt * 16 + l15;
    if (ncol < N) {
        float bv = bias ? bias[ncol] : 0.f;
#pragma unroll
        for (int v = 0; v < 8; ++v) {
            int m = mt * 16 + v + half * 8;      // C/D layout: M = v + 8*half
            float val = acc[v] + bv;
            if (act == 1)      val = silu_f(val);
            else if (act == 2) val = (val > 20.f) ? val : log1pf(__expf(val));
            C[(size_t)m * ldc + ncol] = val;
        }
    }
}

// ---------------------------------------------------------------------------
// f32 (R x K, row stride ld_in) -> f16 (Rpad x Kpad), zero padded.
// rev=1 reverses the L dimension within each batch (rows are b*Lseq + l).
// ---------------------------------------------------------------------------
__global__ void convert_pad_f16(const float* __restrict__ in, int ld_in,
                                int R, int K, _Float16* __restrict__ out,
                                int Kpad, int rev, int Lseq, size_t total) {
    size_t tid = (size_t)blockIdx.x * blockDim.x + threadIdx.x;
    if (tid >= total) return;
    int    k = (int)(tid % (size_t)Kpad);
    size_t r = tid / (size_t)Kpad;
    float v = 0.f;
    if (r < (size_t)R && k < K) {
        size_t sr = r;
        if (rev) {
            size_t b = r / (size_t)Lseq, l = r % (size_t)Lseq;
            sr = b * (size_t)Lseq + (size_t)(Lseq - 1) - l;
        }
        v = in[sr * (size_t)ld_in + k];
    }
    out[r * (size_t)Kpad + k] = (_Float16)v;
}

// ---------------------------------------------------------------------------
// im2col for 8x8/stride-8 patches: x (8,3,224,224) -> A16 (6272,192) f16
// col = c*64 + ki*8 + kj, matching patch_w (E, 3, 8, 8) flattening.
// ---------------------------------------------------------------------------
__global__ void im2col_kernel(const float* __restrict__ x,
                              _Float16* __restrict__ a16, size_t total) {
    size_t tid = (size_t)blockIdx.x * blockDim.x + threadIdx.x;
    if (tid >= total) return;
    int    col = (int)(tid % 192);
    size_t row = tid / 192;
    int b = (int)(row / LSEQ), p = (int)(row % LSEQ);
    int c = col / 64, rem = col % 64, ki = rem / 8, kj = rem % 8;
    int pi = p / 28, pj = p % 28;
    a16[row * 192 + col] =
        (_Float16)x[(((size_t)b * 3 + c) * 224 + pi * 8 + ki) * 224 + pj * 8 + kj];
}

// ---------------------------------------------------------------------------
// Patch post: h[r, 0:192] = silu(LN(pin[r,:])), h[r, 192:196] = rotary concat.
// One block (256 threads) per row r = b*784 + l.
// ---------------------------------------------------------------------------
__global__ void patch_post_kernel(const float* __restrict__ pin,
                                  const float* __restrict__ g,
                                  const float* __restrict__ bb,
                                  float* __restrict__ h) {
    int r = blockIdx.x, c = threadIdx.x;
    __shared__ float sbuf[256];
    float v = (c < 192) ? pin[(size_t)r * 192 + c] : 0.f;
    sbuf[c] = v; __syncthreads();
    for (int s = 128; s > 0; s >>= 1) { if (c < s) sbuf[c] += sbuf[c + s]; __syncthreads(); }
    float mean = sbuf[0] / 192.f; __syncthreads();
    float d = (c < 192) ? (v - mean) : 0.f;
    sbuf[c] = d * d; __syncthreads();
    for (int s = 128; s > 0; s >>= 1) { if (c < s) sbuf[c] += sbuf[c + s]; __syncthreads(); }
    float inv = rsqrtf(sbuf[0] / 192.f + 1e-5f);
    if (c < 192) {
        float y = d * inv * g[c] + bb[c];
        h[(size_t)r * DMODEL + c] = silu_f(y);
    } else if (c < 196) {
        int l = r % LSEQ;
        float scale = 6.28318530717958647692f / 32.f;
        float pos = (float)l * scale;
        float row = (float)(l / 32) * scale;
        float val = (c == 192) ? sinf(pos) : (c == 193) ? cosf(pos)
                  : (c == 194) ? sinf(row) : cosf(row);
        h[(size_t)r * DMODEL + c] = val;
    }
}

// ---------------------------------------------------------------------------
// Depthwise causal conv (width 4) + bias + silu over xz[:, :392].
// Writes u32 (6272,392) f32 and u16 (6272,416) f16 (pad cols zero).
// ---------------------------------------------------------------------------
__global__ void dwconv_kernel(const float* __restrict__ xz,
                              const float* __restrict__ cw,
                              const float* __restrict__ cb,
                              float* __restrict__ u32,
                              _Float16* __restrict__ u16, size_t total) {
    size_t tid = (size_t)blockIdx.x * blockDim.x + threadIdx.x;
    if (tid >= total) return;
    int    dd = (int)(tid % 416);
    size_t r  = tid / 416;
    if (dd >= DIN) { u16[r * 416 + dd] = (_Float16)0.f; return; }
    int l = (int)(r % LSEQ);
    float acc = cb[dd];
#pragma unroll
    for (int k = 0; k < 4; ++k) {
        int tt = l - 3 + k;
        if (tt >= 0) acc += cw[dd * 4 + k] * xz[(r - 3 + k) * 784 + dd];
    }
    float u = silu_f(acc);
    u32[r * DIN + dd] = u;
    u16[r * 416 + dd] = (_Float16)u;
}

// ---------------------------------------------------------------------------
// Selective scan, one thread per (b, d_inner).  16 states in registers.
// Fuses +u*D, *silu(z) gating, and f16 staging for out_proj.
// ---------------------------------------------------------------------------
__global__ void scan_kernel(const float* __restrict__ dt,
                            const float* __restrict__ xdbl,
                            const float* __restrict__ u32,
                            const float* __restrict__ xz,
                            const float* __restrict__ A_log,
                            const float* __restrict__ Dp,
                            _Float16* __restrict__ y16) {
    int tid = blockIdx.x * blockDim.x + threadIdx.x;
    if (tid >= NB * 416) return;
    int b = tid / 416, dd = tid % 416;
    size_t base = (size_t)b * LSEQ;
    if (dd >= DIN) {
        for (int t = 0; t < LSEQ; ++t) y16[(base + t) * 416 + dd] = (_Float16)0.f;
        return;
    }
    float Arow[16], hs[16];
#pragma unroll
    for (int n = 0; n < 16; ++n) { Arow[n] = -__expf(A_log[dd * 16 + n]); hs[n] = 0.f; }
    float Dd = Dp[dd];
    for (int t = 0; t < LSEQ; ++t) {
        size_t r = base + t;
        float dtv = dt[r * DIN + dd];
        float uv  = u32[r * DIN + dd];
        float zv  = xz[r * 784 + DIN + dd];
        const float* xr = xdbl + r * 45;
        float y = 0.f;
#pragma unroll
        for (int n = 0; n < 16; ++n) {
            float dA = __expf(dtv * Arow[n]);
            hs[n] = dA * hs[n] + dtv * xr[13 + n] * uv;
            y += hs[n] * xr[29 + n];
        }
        y = (y + uv * Dd) * silu_f(zv);
        y16[r * 416 + dd] = (_Float16)y;
    }
}

// ---------------------------------------------------------------------------
// o = 0.5*(fwd + rev(bwd)); h += silu(LN(o)).  One block per row.
// ---------------------------------------------------------------------------
__global__ void combine_kernel(const float* __restrict__ fo,
                               const float* __restrict__ bo,
                               const float* __restrict__ g,
                               const float* __restrict__ bb,
                               float* __restrict__ h) {
    int r = blockIdx.x, c = threadIdx.x;
    int b = r / LSEQ, l = r % LSEQ;
    __shared__ float sbuf[256];
    float o = 0.f;
    if (c < DMODEL) {
        float fv = fo[(size_t)r * DMODEL + c];
        float bv = bo[((size_t)b * LSEQ + (LSEQ - 1 - l)) * DMODEL + c];
        o = 0.5f * (fv + bv);
    }
    sbuf[c] = o; __syncthreads();
    for (int s = 128; s > 0; s >>= 1) { if (c < s) sbuf[c] += sbuf[c + s]; __syncthreads(); }
    float mean = sbuf[0] / (float)DMODEL; __syncthreads();
    float d = (c < DMODEL) ? (o - mean) : 0.f;
    sbuf[c] = d * d; __syncthreads();
    for (int s = 128; s > 0; s >>= 1) { if (c < s) sbuf[c] += sbuf[c + s]; __syncthreads(); }
    float inv = rsqrtf(sbuf[0] / (float)DMODEL + 1e-5f);
    if (c < DMODEL) {
        float y = d * inv * g[c] + bb[c];
        h[(size_t)r * DMODEL + c] += silu_f(y);
    }
}

// ---------------------------------------------------------------------------
// Classifier head: out[b] = mean_L(h[b]) . cls_w + cls_b.  One block per b.
// ---------------------------------------------------------------------------
__global__ void head_kernel(const float* __restrict__ h,
                            const float* __restrict__ cw,
                            const float* __restrict__ cb,
                            float* __restrict__ out) {
    int b = blockIdx.x, c = threadIdx.x;
    __shared__ float sbuf[256];
    float s = 0.f;
    if (c < DMODEL) {
        for (int t = 0; t < LSEQ; ++t) s += h[((size_t)b * LSEQ + t) * DMODEL + c];
        s = (s / (float)LSEQ) * cw[c];
    }
    sbuf[c] = s; __syncthreads();
    for (int st = 128; st > 0; st >>= 1) { if (c < st) sbuf[c] += sbuf[c + st]; __syncthreads(); }
    if (c == 0) out[b] = sbuf[0] + cb[0];
}

// ---------------------------------------------------------------------------
// Host side
// ---------------------------------------------------------------------------
struct MParams {
    const float *in_w, *conv_w, *conv_b, *xproj_w, *dt_w, *dt_b, *A_log, *D, *out_w;
};

extern "C" void kernel_launch(void* const* d_in, const int* in_sizes, int n_in,
                              void* d_out, int out_size, void* d_ws, size_t ws_size,
                              hipStream_t stream) {
    (void)n_in; (void)out_size; (void)ws_size;
    auto F = [&](int i) { return (const float*)d_in[i]; };

    const float *x, *patch_w, *patch_b, *pln_g, *pln_b, *cls_w, *cls_b;
    const float *ln_g[2], *ln_b[2];
    MParams mp[2][2]; // [layer][0=fwd,1=bwd]

    if (in_sizes[0] == 1) {
        // alphabetical pytree order: params{cls_b, cls_w, layers, patch_*}, x
        cls_b = F(0); cls_w = F(1);
        int idx = 2;
        for (int L2 = 0; L2 < 2; ++L2) {
            for (int dir = 1; dir >= 0; --dir) { // bwd, fwd
                MParams& m = mp[L2][dir];
                m.A_log = F(idx + 0); m.D     = F(idx + 1); m.conv_b  = F(idx + 2);
                m.conv_w= F(idx + 3); m.dt_b  = F(idx + 4); m.dt_w    = F(idx + 5);
                m.in_w  = F(idx + 6); m.out_w = F(idx + 7); m.xproj_w = F(idx + 8);
                idx += 9;
            }
            ln_b[L2] = F(idx++); ln_g[L2] = F(idx++);
        }
        patch_b = F(idx++); pln_b = F(idx++); pln_g = F(idx++); patch_w = F(idx++);
        x = F(idx++);
    } else {
        // recursive insertion order: x, patch_*, layers{fwd, bwd, ln_g, ln_b}, cls_*
        x = F(0); patch_w = F(1); patch_b = F(2); pln_g = F(3); pln_b = F(4);
        int idx = 5;
        for (int L2 = 0; L2 < 2; ++L2) {
            for (int dir = 0; dir < 2; ++dir) {
                MParams& m = mp[L2][dir];
                m.in_w  = F(idx + 0); m.conv_w = F(idx + 1); m.conv_b  = F(idx + 2);
                m.xproj_w = F(idx + 3); m.dt_w = F(idx + 4); m.dt_b    = F(idx + 5);
                m.A_log = F(idx + 6); m.D      = F(idx + 7); m.out_w   = F(idx + 8);
                idx += 9;
            }
            ln_g[L2] = F(idx++); ln_b[L2] = F(idx++);
        }
        cls_w = F(idx++); cls_b = F(idx++);
    }

    // ---- workspace bump allocator ----
    char* ws = (char*)d_ws;
    size_t off = 0;
    auto alloc = [&](size_t bytes) -> void* {
        size_t o = (off + 255) & ~(size_t)255;
        off = o + bytes;
        return (void*)(ws + o);
    };
    const size_t M = MROWS;
    float* hbuf   = (float*)alloc(M * DMODEL * 4);
    float* fout   = (float*)alloc(M * DMODEL * 4);
    float* bout   = (float*)alloc(M * DMODEL * 4);
    float* xz     = (float*)alloc(M * 784 * 4);
    float* u32    = (float*)alloc(M * DIN * 4);
    float* dtb    = (float*)alloc(M * DIN * 4);   // also patch-GEMM out (stride 192)
    float* xdbl   = (float*)alloc(M * 45 * 4);
    _Float16* h16     = (_Float16*)alloc(M * 224 * 2);
    _Float16* act16   = (_Float16*)alloc(M * 416 * 2); // im2col / u16 / y16
    _Float16* dtlow16 = (_Float16*)alloc(M * 32 * 2);
    _Float16* patchW16 = (_Float16*)alloc((size_t)192 * 192 * 2);
    _Float16 *inW16[2][2], *xpW16[2][2], *dtW16[2][2], *outW16[2][2];
    for (int L2 = 0; L2 < 2; ++L2)
        for (int dir = 0; dir < 2; ++dir) {
            inW16[L2][dir]  = (_Float16*)alloc((size_t)784 * 224 * 2);
            xpW16[L2][dir]  = (_Float16*)alloc((size_t)48 * 416 * 2);
            dtW16[L2][dir]  = (_Float16*)alloc((size_t)400 * 32 * 2);
            outW16[L2][dir] = (_Float16*)alloc((size_t)208 * 416 * 2);
        }

    auto cdiv = [](size_t a, size_t b) { return (unsigned)((a + b - 1) / b); };
    auto wconv = [&](const float* src, int N_, int K_, _Float16* dst, int Npad_, int Kpad_) {
        size_t tot = (size_t)Npad_ * Kpad_;
        convert_pad_f16<<<cdiv(tot, 256), 256, 0, stream>>>(
            src, K_, N_, K_, dst, Kpad_, 0, LSEQ, tot);
    };

    // ---- stage weights to padded f16 ----
    wconv(patch_w, 192, 192, patchW16, 192, 192);
    for (int L2 = 0; L2 < 2; ++L2)
        for (int dir = 0; dir < 2; ++dir) {
            const MParams& m = mp[L2][dir];
            wconv(m.in_w,    784, DMODEL, inW16[L2][dir],  784, 224);
            wconv(m.xproj_w,  45, DIN,    xpW16[L2][dir],   48, 416);
            wconv(m.dt_w,    DIN,  13,    dtW16[L2][dir],  400, 32);
            wconv(m.out_w, DMODEL, DIN,   outW16[L2][dir], 208, 416);
        }

    // ---- patch embed: im2col + WMMA GEMM + LN/SiLU/rotary ----
    {
        size_t tot = M * 192;
        im2col_kernel<<<cdiv(tot, 256), 256, 0, stream>>>(x, act16, tot);
    }
    wmma_gemm_kernel<<<dim3(392, 3), dim3(32, 4), 0, stream>>>(
        act16, patchW16, patch_b, dtb, 192, 192, 12, 192, 0);
    patch_post_kernel<<<(unsigned)M, 256, 0, stream>>>(dtb, pln_g, pln_b, hbuf);

    auto run_mamba = [&](const MParams& m, _Float16* inW, _Float16* xpW,
                         _Float16* dtW, _Float16* outW, float* outC) {
        // in_proj: (6272,224) x (784,224)^T -> xz (6272,784)
        wmma_gemm_kernel<<<dim3(392, 13), dim3(32, 4), 0, stream>>>(
            h16, inW, nullptr, xz, 224, 784, 49, 784, 0);
        // depthwise causal conv + silu
        {
            size_t tot = M * 416;
            dwconv_kernel<<<cdiv(tot, 256), 256, 0, stream>>>(
                xz, m.conv_w, m.conv_b, u32, act16, tot);
        }
        // xproj: (6272,416) x (45,416)^T -> xdbl (6272,45)
        wmma_gemm_kernel<<<dim3(392, 1), dim3(32, 4), 0, stream>>>(
            act16, xpW, nullptr, xdbl, 416, 45, 3, 45, 0);
        // dt low-rank slice -> f16 (K=13 padded to 32)
        {
            size_t tot = M * 32;
            convert_pad_f16<<<cdiv(tot, 256), 256, 0, stream>>>(
                xdbl, 45, (int)M, 13, dtlow16, 32, 0, LSEQ, tot);
        }
        // dt: softplus((6272,32) x (392,32)^T + dt_b) -> dt (6272,392)
        wmma_gemm_kernel<<<dim3(392, 7), dim3(32, 4), 0, stream>>>(
            dtlow16, dtW, m.dt_b, dtb, 32, DIN, 25, DIN, 2);
        // selective scan (fused gating), writes y16 into act16
        scan_kernel<<<cdiv((size_t)NB * 416, 256), 256, 0, stream>>>(
            dtb, xdbl, u32, xz, m.A_log, m.D, act16);
        // out_proj: (6272,416) x (196,416)^T -> outC (6272,196)
        wmma_gemm_kernel<<<dim3(392, 4), dim3(32, 4), 0, stream>>>(
            act16, outW, nullptr, outC, 416, DMODEL, 13, DMODEL, 0);
    };

    for (int L2 = 0; L2 < 2; ++L2) {
        size_t tot = M * 224;
        convert_pad_f16<<<cdiv(tot, 256), 256, 0, stream>>>(
            hbuf, DMODEL, (int)M, DMODEL, h16, 224, 0, LSEQ, tot);
        run_mamba(mp[L2][0], inW16[L2][0], xpW16[L2][0], dtW16[L2][0], outW16[L2][0], fout);
        convert_pad_f16<<<cdiv(tot, 256), 256, 0, stream>>>(
            hbuf, DMODEL, (int)M, DMODEL, h16, 224, 1, LSEQ, tot);
        run_mamba(mp[L2][1], inW16[L2][1], xpW16[L2][1], dtW16[L2][1], outW16[L2][1], bout);
        combine_kernel<<<(unsigned)M, 256, 0, stream>>>(fout, bout, ln_g[L2], ln_b[L2], hbuf);
    }

    head_kernel<<<NB, 256, 0, stream>>>(hbuf, cls_w, cls_b, (float*)d_out);
}
